// MultiHeadAttentionQuantum_65481071398657
// MI455X (gfx1250) — matrix operand order repaired and checked by reading, not compiled, not run on hardware
//
#include <hip/hip_runtime.h>

// ---- problem constants -------------------------------------------------
#define SLEN   2048
#define EDIM   1024
#define HEADS  16
#define DHEAD  64
#define MROWS  4096   // B*S
#define KDIM   1024

typedef __attribute__((ext_vector_type(16))) __bf16 v16bf;
typedef __attribute__((ext_vector_type(8)))  float  v8f;
typedef __attribute__((ext_vector_type(4)))  int    v4i;

union Frag {           // A or B operand of v_wmma_f32_16x16x32_bf16
  v16bf v;
  uint4 q[2];
};
union CAcc {           // C/D accumulator
  v8f   v;
  float f[8];
};

__device__ __forceinline__ unsigned short f2bf(float f) {
  unsigned int u = __float_as_uint(f);
  u += 0x7FFFu + ((u >> 16) & 1u);   // round-to-nearest-even
  return (unsigned short)(u >> 16);
}

// ---- optional CDNA5 async global->LDS copies (guarded; fallback kept) --
#if defined(__has_builtin)
#if __has_builtin(__builtin_amdgcn_global_load_async_to_lds_b128) && \
    __has_builtin(__builtin_amdgcn_s_wait_asynccnt)
#define USE_ASYNC_LDS 1
#endif
#endif
#ifndef USE_ASYNC_LDS
#define USE_ASYNC_LDS 0
#endif

#if USE_ASYNC_LDS
// signature (from hipcc diagnostic): arg0 = AS(1) v4i* (non-const), 4 args
typedef __attribute__((address_space(1))) v4i gv4i;
typedef __attribute__((address_space(3))) v4i lv4i;
__device__ __forceinline__ void async_cp16(unsigned short* lds_dst,
                                           const unsigned short* gsrc) {
  __builtin_amdgcn_global_load_async_to_lds_b128(
      (gv4i*)gsrc, (lv4i*)lds_dst, 0, 0);
}
#endif

// ---- K0: fp32 -> bf16 --------------------------------------------------
__global__ void convert_bf16(const float* __restrict__ src,
                             unsigned short* __restrict__ dst, int n) {
  int i = (blockIdx.x * blockDim.x + threadIdx.x) * 4;
  if (i >= n) return;
  float4 f = *reinterpret_cast<const float4*>(src + i);
  uint2 p;
  p.x = (unsigned)f2bf(f.x) | ((unsigned)f2bf(f.y) << 16);
  p.y = (unsigned)f2bf(f.z) | ((unsigned)f2bf(f.w) << 16);
  *reinterpret_cast<uint2*>(dst + i) = p;
}

// LDS row stride (ushorts) for 64-wide bf16 tiles: 144B rows, 16B aligned,
// odd multiple of 16B -> spreads banks across rows.
#define LK 72

// Shared GEMM stage: workgroup tile 128(M) x 64(N), K-chunk 64.
// Per wave: 32x32 output = 4 accumulators, 8 WMMAs per stage.
// lA: [128][LK] bf16 rows of X ; lB: [64][LK] bf16 rows of W (output cols)
__device__ __forceinline__ void stage_tiles(
    unsigned short* lA, unsigned short* lB,
    const unsigned short* __restrict__ gA, const unsigned short* __restrict__ gB,
    int m0, int n0, int kc, int ldg, int tid) {
  const int ar = tid >> 1;            // 0..127
  const int ac = (tid & 1) * 32;      // 0 / 32
  const int br = tid >> 2;            // 0..63
  const int bc = (tid & 3) * 16;      // 0/16/32/48
#if USE_ASYNC_LDS
  for (int j = 0; j < 4; ++j)
    async_cp16(&lA[ar * LK + ac + j * 8],
               &gA[(size_t)(m0 + ar) * ldg + kc + ac + j * 8]);
  for (int j = 0; j < 2; ++j)
    async_cp16(&lB[br * LK + bc + j * 8],
               &gB[(size_t)(n0 + br) * ldg + kc + bc + j * 8]);
  __builtin_amdgcn_s_wait_asynccnt(0);
#else
  for (int j = 0; j < 4; ++j)
    *reinterpret_cast<uint4*>(&lA[ar * LK + ac + j * 8]) =
        *reinterpret_cast<const uint4*>(&gA[(size_t)(m0 + ar) * ldg + kc + ac + j * 8]);
  for (int j = 0; j < 2; ++j)
    *reinterpret_cast<uint4*>(&lB[br * LK + bc + j * 8]) =
        *reinterpret_cast<const uint4*>(&gB[(size_t)(n0 + br) * ldg + kc + bc + j * 8]);
#endif
  if (kc + 64 < KDIM) {               // speculative prefetch of next K stage
    __builtin_prefetch(&gA[(size_t)(m0 + ar) * ldg + kc + 64 + ac], 0, 3);
    __builtin_prefetch(&gB[(size_t)(n0 + br) * ldg + kc + 64 + bc], 0, 3);
  }
}

// One K=64 stage of WMMAs for a 32x32 per-wave tile.
__device__ __forceinline__ void stage_mma(
    const unsigned short* lA, const unsigned short* lB,
    int m_off, int n_off, int half, int l16,
    CAcc& c00, CAcc& c01, CAcc& c10, CAcc& c11) {
#pragma unroll
  for (int c = 0; c < 64; c += 32) {
    Frag a0, a1, b0, b1;
    {   // A 16x32: lanes0-15 row=l16 K0..7|16..23 ; lanes16-31 K8..15|24..31
      const int kb = c + half * 8;
      a0.q[0] = *reinterpret_cast<const uint4*>(&lA[(m_off + l16) * LK + kb]);
      a0.q[1] = *reinterpret_cast<const uint4*>(&lA[(m_off + l16) * LK + kb + 16]);
      a1.q[0] = *reinterpret_cast<const uint4*>(&lA[(m_off + 16 + l16) * LK + kb]);
      a1.q[1] = *reinterpret_cast<const uint4*>(&lA[(m_off + 16 + l16) * LK + kb + 16]);
    }
    {   // B 32x16: lane holds one output column, K 8h..8h+7 | 16+8h..16+8h+7
      const int kb = c + half * 8;
      b0.q[0] = *reinterpret_cast<const uint4*>(&lB[(n_off + l16) * LK + kb]);
      b0.q[1] = *reinterpret_cast<const uint4*>(&lB[(n_off + l16) * LK + kb + 16]);
      b1.q[0] = *reinterpret_cast<const uint4*>(&lB[(n_off + 16 + l16) * LK + kb]);
      b1.q[1] = *reinterpret_cast<const uint4*>(&lB[(n_off + 16 + l16) * LK + kb + 16]);
    }
    c00.v = __builtin_amdgcn_wmma_f32_16x16x32_bf16(false, a0.v, false, b0.v,
                                                    (short)0, c00.v, false, false);
    c01.v = __builtin_amdgcn_wmma_f32_16x16x32_bf16(false, a0.v, false, b1.v,
                                                    (short)0, c01.v, false, false);
    c10.v = __builtin_amdgcn_wmma_f32_16x16x32_bf16(false, a1.v, false, b0.v,
                                                    (short)0, c10.v, false, false);
    c11.v = __builtin_amdgcn_wmma_f32_16x16x32_bf16(false, a1.v, false, b1.v,
                                                    (short)0, c11.v, false, false);
  }
}

// ---- K1: fused QKV projection + quantum (cumprod(cos)) -----------------
// z=0: Q -> [BH,S,D]; z=1: K -> [BH,S,D]; z=2: V -> transposed [BH,D,S]
__global__ __launch_bounds__(256) void proj_quantum(
    const unsigned short* __restrict__ xb,   // [4096,1024] bf16
    const unsigned short* __restrict__ wq,
    const unsigned short* __restrict__ wk,
    const unsigned short* __restrict__ wv,   // [1024,1024] bf16 (row n = out col)
    unsigned short* __restrict__ qout,
    unsigned short* __restrict__ kout,
    unsigned short* __restrict__ vtout) {
  const int sel = blockIdx.z;
  const unsigned short* __restrict__ W = (sel == 0) ? wq : ((sel == 1) ? wk : wv);
  const int m0 = blockIdx.x * 128;
  const int h  = blockIdx.y;
  const int n0 = h * DHEAD;

  // staging (27.6KB) and fp32 C tile (33.3KB) overlap in one allocation
  __shared__ __align__(16) unsigned char smem[128 * 65 * 4];
  unsigned short* lA = reinterpret_cast<unsigned short*>(smem);
  unsigned short* lB = lA + 128 * LK;
  float*          lC = reinterpret_cast<float*>(smem);

  const int tid  = threadIdx.x;
  const int wave = tid >> 5;
  const int lane = tid & 31;
  const int half = lane >> 4;
  const int l16  = lane & 15;
  const int m_off = (wave >> 1) * 32;
  const int n_off = (wave & 1) * 32;

  CAcc c00, c01, c10, c11;
  c00.v = (v8f){}; c01.v = (v8f){}; c10.v = (v8f){}; c11.v = (v8f){};

  for (int kc = 0; kc < KDIM; kc += 64) {
    __syncthreads();
    stage_tiles(lA, lB, xb, W, m0, n0, kc, KDIM, tid);
    __syncthreads();
    stage_mma(lA, lB, m_off, n_off, half, l16, c00, c01, c10, c11);
  }

  __syncthreads();                    // staging dead; reuse smem as lC
  for (int r = 0; r < 8; ++r) {       // C layout: VGPR r = row r (+8 hi half)
    const int m = m_off + r + half * 8;
    lC[m * 65 + n_off + l16]             = c00.f[r];
    lC[m * 65 + n_off + 16 + l16]        = c01.f[r];
    lC[(m + 16) * 65 + n_off + l16]      = c10.f[r];
    lC[(m + 16) * 65 + n_off + 16 + l16] = c11.f[r];
  }
  __syncthreads();

  if (tid < 128) {                    // quantum: cumprod(cos) along head dim
    const int mg = m0 + tid;
    const int b  = mg >> 11;          // / 2048
    const int s  = mg & 2047;
    const int bh = b * HEADS + h;
    float prod = 1.0f;
    if (sel == 2) {
      for (int d = 0; d < DHEAD; ++d) {
        prod *= __cosf(lC[tid * 65 + d]);
        vtout[((size_t)bh * DHEAD + d) * SLEN + s] = f2bf(prod);
      }
    } else {
      unsigned short* __restrict__ dst = (sel == 0) ? qout : kout;
      for (int d = 0; d < DHEAD; ++d) {
        prod *= __cosf(lC[tid * 65 + d]);
        dst[((size_t)bh * SLEN + s) * DHEAD + d] = f2bf(prod);
      }
    }
  }
}

// ---- K2: flash attention (online softmax, bf16 WMMA) -------------------
#define LSTR 40
__global__ __launch_bounds__(128) void flash_attn(
    const unsigned short* __restrict__ qb,   // [BH,S,D]
    const unsigned short* __restrict__ kb,   // [BH,S,D]
    const unsigned short* __restrict__ vt,   // [BH,D,S] (pre-transposed)
    unsigned short* __restrict__ attnb) {    // [B,S,E] bf16
  const int bh   = blockIdx.y;
  const int b    = bh >> 4;
  const int h    = bh & 15;
  const int tid  = threadIdx.x;
  const int wave = tid >> 5;
  const int lane = tid & 31;
  const int half = lane >> 4;
  const int l16  = lane & 15;
  const int row0 = blockIdx.x * 64 + wave * 16;   // this wave's query block

  __shared__ __align__(16) unsigned short lP[4][16 * LSTR];

  const unsigned short* __restrict__ qp = qb + ((size_t)bh * SLEN + row0) * DHEAD;
  const unsigned short* __restrict__ kp = kb + (size_t)bh * SLEN * DHEAD;
  const unsigned short* __restrict__ vp = vt + (size_t)bh * DHEAD * SLEN;

  Frag aq[2];                        // Q 16x64 as two 16x32 A fragments
  {
    const int kbse = half * 8;
    for (int c = 0; c < 2; ++c) {
      const int base = l16 * DHEAD + c * 32 + kbse;
      aq[c].q[0] = *reinterpret_cast<const uint4*>(&qp[base]);
      aq[c].q[1] = *reinterpret_cast<const uint4*>(&qp[base + 16]);
    }
  }

  CAcc acc[4];
  for (int t = 0; t < 4; ++t) acc[t].v = (v8f){};
  float mrun[8], lrun[8];
  for (int r = 0; r < 8; ++r) { mrun[r] = -3.0e38f; lrun[r] = 0.0f; }

  for (int kblk = 0; kblk < SLEN; kblk += 32) {
    // ---- scores S = Q.K^T (16x32), K=64 chained over two 32-chunks ----
    CAcc s0, s1;
    s0.v = (v8f){}; s1.v = (v8f){};
    for (int c = 0; c < 2; ++c) {
      Frag bk0, bk1;
      const int dof = c * 32 + half * 8;
      bk0.q[0] = *reinterpret_cast<const uint4*>(&kp[(size_t)(kblk + l16) * DHEAD + dof]);
      bk0.q[1] = *reinterpret_cast<const uint4*>(&kp[(size_t)(kblk + l16) * DHEAD + dof + 16]);
      bk1.q[0] = *reinterpret_cast<const uint4*>(&kp[(size_t)(kblk + 16 + l16) * DHEAD + dof]);
      bk1.q[1] = *reinterpret_cast<const uint4*>(&kp[(size_t)(kblk + 16 + l16) * DHEAD + dof + 16]);
      s0.v = __builtin_amdgcn_wmma_f32_16x16x32_bf16(false, aq[c].v, false, bk0.v,
                                                     (short)0, s0.v, false, false);
      s1.v = __builtin_amdgcn_wmma_f32_16x16x32_bf16(false, aq[c].v, false, bk1.v,
                                                     (short)0, s1.v, false, false);
    }
    // ---- online softmax: rows live in 16-lane groups of one VGPR ----
    for (int r = 0; r < 8; ++r) {
      float a0 = s0.f[r] * 0.125f;           // 1/sqrt(64)
      float a1 = s1.f[r] * 0.125f;
      float mx = fmaxf(a0, a1);
      for (int mk = 1; mk < 16; mk <<= 1)
        mx = fmaxf(mx, __shfl_xor(mx, mk, 32));
      const float mnew = fmaxf(mrun[r], mx);
      const float corr = __expf(mrun[r] - mnew);
      const float p0 = __expf(a0 - mnew);
      const float p1 = __expf(a1 - mnew);
      float rs = p0 + p1;
      for (int mk = 1; mk < 16; mk <<= 1)
        rs += __shfl_xor(rs, mk, 32);
      lrun[r] = lrun[r] * corr + rs;
      mrun[r] = mnew;
      for (int t = 0; t < 4; ++t) acc[t].f[r] *= corr;
      const int prow = r + half * 8;         // re-layout P: C-frag -> LDS
      lP[wave][prow * LSTR + l16]      = f2bf(p0);
      lP[wave][prow * LSTR + 16 + l16] = f2bf(p1);
    }
    __syncthreads();
    // ---- O += P.V : A = P (16x32) from LDS, B = Vt (contiguous keys) ----
    Frag ap;
    {
      const int kbse = half * 8;
      ap.q[0] = *reinterpret_cast<const uint4*>(&lP[wave][l16 * LSTR + kbse]);
      ap.q[1] = *reinterpret_cast<const uint4*>(&lP[wave][l16 * LSTR + 16 + kbse]);
    }
    for (int t = 0; t < 4; ++t) {
      Frag bv;
      const int d   = t * 16 + l16;
      const int kof = kblk + half * 8;
      bv.q[0] = *reinterpret_cast<const uint4*>(&vp[(size_t)d * SLEN + kof]);
      bv.q[1] = *reinterpret_cast<const uint4*>(&vp[(size_t)d * SLEN + kof + 16]);
      acc[t].v = __builtin_amdgcn_wmma_f32_16x16x32_bf16(false, ap.v, false, bv.v,
                                                         (short)0, acc[t].v, false, false);
    }
    __syncthreads();
  }

  // ---- epilogue: O / l, write bf16 to [B,S,H*D] -----------------------
  for (int r = 0; r < 8; ++r) {
    const float inv = 1.0f / lrun[r];
    const int s = row0 + r + half * 8;
    for (int t = 0; t < 4; ++t) {
      const int e = h * DHEAD + t * 16 + l16;
      attnb[((size_t)b * SLEN + s) * EDIM + e] = f2bf(acc[t].f[r] * inv);
    }
  }
}

// ---- K3: output projection + bias (fp32 out) ---------------------------
__global__ __launch_bounds__(256) void out_proj(
    const unsigned short* __restrict__ ab,   // [4096,1024] bf16
    const unsigned short* __restrict__ wc,   // [1024,1024] bf16
    const float* __restrict__ bc,
    float* __restrict__ out) {
  const int m0 = blockIdx.x * 128;
  const int n0 = blockIdx.y * 64;
  __shared__ __align__(16) unsigned short lA[128 * LK];
  __shared__ __align__(16) unsigned short lB[64 * LK];
  const int tid  = threadIdx.x;
  const int wave = tid >> 5;
  const int lane = tid & 31;
  const int half = lane >> 4;
  const int l16  = lane & 15;
  const int m_off = (wave >> 1) * 32;
  const int n_off = (wave & 1) * 32;

  CAcc c00, c01, c10, c11;
  c00.v = (v8f){}; c01.v = (v8f){}; c10.v = (v8f){}; c11.v = (v8f){};

  for (int kc = 0; kc < EDIM; kc += 64) {
    __syncthreads();
    stage_tiles(lA, lB, ab, wc, m0, n0, kc, EDIM, tid);
    __syncthreads();
    stage_mma(lA, lB, m_off, n_off, half, l16, c00, c01, c10, c11);
  }

  const float bias0 = bc[n0 + n_off + l16];
  const float bias1 = bc[n0 + n_off + 16 + l16];
  for (int r = 0; r < 8; ++r) {
    const int m = m0 + m_off + r + half * 8;
    out[(size_t)m * EDIM + n0 + n_off + l16]             = c00.f[r] + bias0;
    out[(size_t)m * EDIM + n0 + n_off + 16 + l16]        = c01.f[r] + bias1;
    out[(size_t)(m + 16) * EDIM + n0 + n_off + l16]      = c10.f[r] + bias0;
    out[(size_t)(m + 16) * EDIM + n0 + n_off + 16 + l16] = c11.f[r] + bias1;
  }
}

// ---- host-side launch --------------------------------------------------
extern "C" void kernel_launch(void* const* d_in, const int* in_sizes, int n_in,
                              void* d_out, int out_size, void* d_ws, size_t ws_size,
                              hipStream_t stream) {
  const float* x  = (const float*)d_in[0];
  const float* Wq = (const float*)d_in[1];
  const float* Wk = (const float*)d_in[2];
  const float* Wv = (const float*)d_in[3];
  const float* Wc = (const float*)d_in[4];
  const float* bc = (const float*)d_in[5];

  unsigned short* ws    = (unsigned short*)d_ws;
  unsigned short* xb    = ws;                                   // 4096*1024
  unsigned short* wqb   = xb    + (size_t)MROWS * KDIM;
  unsigned short* wkb   = wqb   + (size_t)EDIM * EDIM;
  unsigned short* wvb   = wkb   + (size_t)EDIM * EDIM;
  unsigned short* wcb   = wvb   + (size_t)EDIM * EDIM;
  unsigned short* qbuf  = wcb   + (size_t)EDIM * EDIM;          // [32,2048,64]
  unsigned short* kbuf  = qbuf  + (size_t)32 * SLEN * DHEAD;
  unsigned short* vtbuf = kbuf  + (size_t)32 * SLEN * DHEAD;    // [32,64,2048]
  unsigned short* abuf  = vtbuf + (size_t)32 * SLEN * DHEAD;    // [4096,1024]

  convert_bf16<<<4096, 256, 0, stream>>>(x,  xb,  MROWS * KDIM);
  convert_bf16<<<1024, 256, 0, stream>>>(Wq, wqb, EDIM * EDIM);
  convert_bf16<<<1024, 256, 0, stream>>>(Wk, wkb, EDIM * EDIM);
  convert_bf16<<<1024, 256, 0, stream>>>(Wv, wvb, EDIM * EDIM);
  convert_bf16<<<1024, 256, 0, stream>>>(Wc, wcb, EDIM * EDIM);

  proj_quantum<<<dim3(32, 16, 3), 256, 0, stream>>>(xb, wqb, wkb, wvb,
                                                    qbuf, kbuf, vtbuf);
  flash_attn<<<dim3(32, 32), 128, 0, stream>>>(qbuf, kbuf, vtbuf, abuf);
  out_proj<<<dim3(32, 16), 256, 0, stream>>>(abuf, wcb, bc, (float*)d_out);
}
